// SelfAttention_37666863186591
// MI455X (gfx1250) — matrix operand order repaired and checked by reading, not compiled, not run on hardware
//
#include <hip/hip_runtime.h>

#define EMBED 1024
#define HEADS 16
#define HS 64
#define BATCH 4
#define SEQ 2048
#define C3 (3 * EMBED)

typedef __attribute__((ext_vector_type(16))) __bf16 v16bf;
typedef __attribute__((ext_vector_type(8)))  __bf16 v8bf;
typedef __attribute__((ext_vector_type(8)))  float  v8f;
typedef __attribute__((ext_vector_type(4)))  unsigned int u32x4;
typedef __attribute__((ext_vector_type(4)))  int i32x4;
typedef __attribute__((ext_vector_type(8)))  int i32x8;

__device__ __forceinline__ v8f wmma_bf16(v16bf a, v16bf b, v8f c) {
  return __builtin_amdgcn_wmma_f32_16x16x32_bf16(false, a, false, b, (short)0, c,
                                                 false, false);
}

// 16 contiguous bf16 -> fragment (two 16B loads, register concat).
__device__ __forceinline__ v16bf ld_frag16(const __bf16* p) {
  v8bf a = *(const v8bf*)p;
  v8bf b = *(const v8bf*)(p + 8);
  return __builtin_shufflevector(a, b, 0, 1, 2, 3, 4, 5, 6, 7, 8, 9, 10, 11, 12,
                                 13, 14, 15);
}

// A-operand (16x32): lane<16: row=lane, K=[0..15]; lane>=16: row=lane-16,
// K=[16..31].  ka = k0+hi*16, kb = k0+hi*16+8.
__device__ __forceinline__ v16bf ld_frag_split(const __bf16* p, int ka, int kb) {
  v8bf a = *(const v8bf*)(p + ka);
  v8bf b = *(const v8bf*)(p + kb);
  return __builtin_shufflevector(a, b, 0, 1, 2, 3, 4, 5, 6, 7, 8, 9, 10, 11, 12,
                                 13, 14, 15);
}

// ---- Tensor Data Mover: 2-D bf16 tile -> LDS (D# per CDNA5 ISA ch.8) -------
// data_size=2B; optional LDS padding (pad codes per ISA: interval 2^(c+1)
// dwords, amount (c+1) dwords) so staged rows get +16B stride for bank spread.
__device__ __forceinline__ void tdm_load_2d_bf16(
    unsigned lds_byte_off, const void* gptr, unsigned tile_d0 /*elems, contig*/,
    unsigned tile_d1 /*rows*/, unsigned long long row_stride_elems,
    unsigned tensor_d1 /*rows avail (OOB bound)*/, unsigned pad_interval_code,
    unsigned pad_amount_code) {
  unsigned long long ga = (unsigned long long)gptr;
  u32x4 g0;
  g0[0] = 1u;                                   // count=1, user descriptor
  g0[1] = lds_byte_off;                         // lds_addr (bytes)
  g0[2] = (unsigned)ga;                         // global_addr[31:0]
  g0[3] = (unsigned)((ga >> 32) & 0x01FFFFFFu)  // global_addr[56:32]
          | (2u << 30);                         // type=2 ("image")
  i32x8 g1;
  g1[0] = (int)((1u << 16) | (1u << 20) |       // data_size=2B, pad_enable
                (pad_interval_code << 22) | (pad_amount_code << 25));
  g1[1] = (int)((tile_d0 & 0xFFFFu) << 16);     // tensor_dim0[15:0] (=tile_d0)
  g1[2] = (int)((tile_d0 >> 16) | ((tensor_d1 & 0xFFFFu) << 16));
  g1[3] = (int)((tensor_d1 >> 16) | (tile_d0 << 16));  // tile_dim0
  g1[4] = (int)(tile_d1 & 0xFFFFu);             // tile_dim1; tile_dim2=0
  g1[5] = (int)(unsigned)(row_stride_elems & 0xFFFFFFFFull);
  g1[6] = (int)(unsigned)((row_stride_elems >> 32) & 0xFFFFull);
  g1[7] = 0;
  i32x4 z4 = {0, 0, 0, 0};
  i32x8 z8 = {0, 0, 0, 0, 0, 0, 0, 0};
  // 6-arg toolchain variant: (g0, g1, g2, g3, g4, cpol)
  __builtin_amdgcn_tensor_load_to_lds(g0, g1, z4, z4, z8, 0);
}

__device__ __forceinline__ unsigned lds_off(const void* p) {
  // flat LDS addresses carry the byte offset in addr[31:0]
  return (unsigned)(unsigned long long)p;
}

// ---------------- one-time fp32 -> bf16 conversion --------------------------
__global__ __launch_bounds__(256) void f32_to_bf16(const float* __restrict__ in,
                                                   __bf16* __restrict__ out,
                                                   int n) {
  int i = (blockIdx.x * 256 + threadIdx.x) * 8;
  if (i >= n) return;
  float4 a = *(const float4*)(in + i);
  float4 b = *(const float4*)(in + i + 4);
  v8bf o;
  o[0] = (__bf16)a.x; o[1] = (__bf16)a.y; o[2] = (__bf16)a.z; o[3] = (__bf16)a.w;
  o[4] = (__bf16)b.x; o[5] = (__bf16)b.y; o[6] = (__bf16)b.z; o[7] = (__bf16)b.w;
  *(v8bf*)(out + i) = o;
}

// ---------------- V transpose: vt[b,h,d,t] = qkv_v[b,t,h,d] ----------------
__global__ __launch_bounds__(256) void transpose_v(const __bf16* __restrict__ qkvb,
                                                   __bf16* __restrict__ vt) {
  size_t tid = (size_t)blockIdx.x * 256 + threadIdx.x;
  int t = (int)(tid % SEQ);
  size_t r = tid / SEQ;
  int d = (int)(r % HS);
  size_t r2 = r / HS;
  int h = (int)(r2 % HEADS);
  int b = (int)(r2 / HEADS);
  vt[tid] = qkvb[((size_t)b * SEQ + t) * C3 + 2 * EMBED + h * HS + d];
}

// ---- Out[M,N] = A[M,K] @ W[N,K]^T + bias. Wave: 32(M)x64(N); 8 waves/block.
template <typename OutT>
__global__ __launch_bounds__(256) void gemm_bias_wmma(
    const __bf16* __restrict__ A, const __bf16* __restrict__ W,
    const float* __restrict__ bias, OutT* __restrict__ Out, int N, int K) {
  const int lane = threadIdx.x & 31;
  const int wave = threadIdx.x >> 5;
  const int hi   = lane >> 4;
  const int ln   = lane & 15;
  const int M0   = blockIdx.y * 64 + (wave & 1) * 32;
  const int N0   = blockIdx.x * 256 + (wave >> 1) * 64;

  v8f acc[2][4] = {};
  const __bf16* arow0 = A + (size_t)(M0 + ln) * K;
  const __bf16* arow1 = arow0 + (size_t)16 * K;
  const __bf16* wrow  = W + (size_t)(N0 + ln) * K;
  for (int k = 0; k < K; k += 32) {
    const int ka = k + hi * 16, kb = k + hi * 16 + 8;
    v16bf a0 = ld_frag_split(arow0, ka, kb);
    v16bf a1 = ld_frag_split(arow1, ka, kb);
#pragma unroll
    for (int t = 0; t < 4; ++t) {
      v16bf b = ld_frag16(wrow + (size_t)(t * 16) * K + k + hi * 16);
      acc[0][t] = wmma_bf16(a0, b, acc[0][t]);
      acc[1][t] = wmma_bf16(a1, b, acc[1][t]);
    }
  }
#pragma unroll
  for (int u = 0; u < 2; ++u)
#pragma unroll
    for (int r = 0; r < 8; ++r) {
      OutT* crow = Out + (size_t)(M0 + u * 16 + r + hi * 8) * N + N0 + ln;
#pragma unroll
      for (int t = 0; t < 4; ++t)
        crow[t * 16] = (OutT)(acc[u][t][r] + bias[N0 + t * 16 + ln]);
    }
}

// ---- Flash attention: TDM-staged K/V tiles in LDS, shared by 8 waves -------
__global__ __launch_bounds__(256) void attn_wmma(const __bf16* __restrict__ qkvb,
                                                 const __bf16* __restrict__ vt,
                                                 __bf16* __restrict__ y) {
  const int lane = threadIdx.x & 31;
  const int wave = threadIdx.x >> 5;
  const int hi   = lane >> 4;
  const int ln   = lane & 15;
  const int b    = blockIdx.z;
  const int h    = blockIdx.y;
  const int Q0   = blockIdx.x * 128;       // block q range [Q0, Q0+128)
  const int q0   = Q0 + wave * 16;         // this wave's 16-row q tile

  __shared__ alignas(16) __bf16 sK[32][72];     // 32 keys x 64 d (+16B pad)
  __shared__ alignas(16) __bf16 sV[64][40];     // 64 d x 32 keys (+16B pad)
  __shared__ alignas(16) __bf16 pbuf[8][16][40];
  __bf16 (*P)[40] = pbuf[wave];

  const __bf16* base  = qkvb + (size_t)b * SEQ * C3;
  const __bf16* vhead = vt + ((size_t)b * HEADS + h) * HS * SEQ;  // [d][t]

  const __bf16* qrow = base + (size_t)(q0 + ln) * C3 + h * HS;
  const v16bf aq0 = ld_frag_split(qrow, hi * 16, 8 + hi * 16);
  const v16bf aq1 = ld_frag_split(qrow, 32 + hi * 16, 40 + hi * 16);

  v8f o[4] = {};
  float rmax[8], rsum[8];
#pragma unroll
  for (int r = 0; r < 8; ++r) { rmax[r] = -__builtin_inff(); rsum[r] = 0.0f; }

  const float scale = 0.125f;  // 1/sqrt(64)
  const int kend = q0 + 16;
  const int kmax = Q0 + 128;   // uniform trip count for the whole block

  for (int kk = 0; kk < kmax; kk += 32) {
    if (wave == 0) {
      // K chunk: 32 rows (keys) x 64 elems, global row stride C3
      tdm_load_2d_bf16(lds_off(sK), base + (size_t)kk * C3 + EMBED + h * HS,
                       64, 32, C3, (unsigned)(SEQ - kk), 4 /*32dw*/, 3 /*4dw*/);
      // V^T chunk: 64 rows (d) x 32 elems (keys), global row stride SEQ
      tdm_load_2d_bf16(lds_off(sV), vhead + kk, 32, 64, SEQ, HS,
                       3 /*16dw*/, 3 /*4dw*/);
      __builtin_amdgcn_s_wait_tensorcnt(0);
    }
    __syncthreads();  // staged tiles visible to all waves

    if (kk < kend) {  // uniform per wave (causal skip)
      // ---- S = Q K^T from LDS ----
      v8f s0 = {}, s1 = {};
      s0 = wmma_bf16(aq0, ld_frag16(&sK[ln][hi * 16]), s0);
      s0 = wmma_bf16(aq1, ld_frag16(&sK[ln][32 + hi * 16]), s0);
      s1 = wmma_bf16(aq0, ld_frag16(&sK[16 + ln][hi * 16]), s1);
      s1 = wmma_bf16(aq1, ld_frag16(&sK[16 + ln][32 + hi * 16]), s1);

      // ---- scale, causal mask, online softmax ----
      float corr[8];
#pragma unroll
      for (int r = 0; r < 8; ++r) {
        const int q = q0 + r + hi * 8;
        const float v0 = (kk + ln      <= q) ? s0[r] * scale : -__builtin_inff();
        const float v1 = (kk + 16 + ln <= q) ? s1[r] * scale : -__builtin_inff();
        float m = fmaxf(v0, v1);
#pragma unroll
        for (int off = 1; off < 16; off <<= 1) m = fmaxf(m, __shfl_xor(m, off, 32));
        const float nmax = fmaxf(rmax[r], m);
        const float p0 = __expf(v0 - nmax);
        const float p1 = __expf(v1 - nmax);
        float rs = p0 + p1;
#pragma unroll
        for (int off = 1; off < 16; off <<= 1) rs += __shfl_xor(rs, off, 32);
        corr[r] = __expf(rmax[r] - nmax);
        rsum[r] = rsum[r] * corr[r] + rs;
        rmax[r] = nmax;
        const int mrow = r + hi * 8;
        P[mrow][ln]      = (__bf16)p0;
        P[mrow][16 + ln] = (__bf16)p1;
      }
#pragma unroll
      for (int t = 0; t < 4; ++t)
#pragma unroll
        for (int r = 0; r < 8; ++r) o[t][r] *= corr[r];

      // same-wave LDS round trip for P re-layout (DS in-order per wave)
      asm volatile("s_wait_dscnt 0" ::: "memory");
      v16bf pa = ld_frag_split(&P[ln][0], hi * 16, 8 + hi * 16);

      // ---- O += P V from LDS ----
#pragma unroll
      for (int t = 0; t < 4; ++t) {
        v16bf bv = ld_frag16(&sV[t * 16 + ln][hi * 16]);
        o[t] = wmma_bf16(pa, bv, o[t]);
      }
    }
    __syncthreads();  // protect sK/sV before next chunk overwrites
  }

  // ---- epilogue: normalize, write bf16 y[b, q, h*64 + d] ----
#pragma unroll
  for (int r = 0; r < 8; ++r) {
    const int m = q0 + r + hi * 8;
    const float inv = 1.0f / rsum[r];
    __bf16* yp = y + ((size_t)b * SEQ + m) * EMBED + h * HS;
#pragma unroll
    for (int t = 0; t < 4; ++t) yp[t * 16 + ln] = (__bf16)(o[t][r] * inv);
  }
}

extern "C" void kernel_launch(void* const* d_in, const int* in_sizes, int n_in,
                              void* d_out, int out_size, void* d_ws, size_t ws_size,
                              hipStream_t stream) {
  (void)in_sizes; (void)n_in; (void)out_size; (void)ws_size;
  const float* x      = (const float*)d_in[0];
  const float* w_attn = (const float*)d_in[1];
  const float* b_attn = (const float*)d_in[2];
  const float* w_proj = (const float*)d_in[3];
  const float* b_proj = (const float*)d_in[4];
  float* out = (float*)d_out;

  const int M = BATCH * SEQ;  // 8192

  __bf16* xb   = (__bf16*)d_ws;                           // 16 MB
  __bf16* wab  = xb + (size_t)M * EMBED;                  //  6 MB
  __bf16* wpb  = wab + (size_t)C3 * EMBED;                //  2 MB
  __bf16* qkvb = wpb + (size_t)EMBED * EMBED;             // 48 MB
  __bf16* vtb  = qkvb + (size_t)M * C3;                   // 16 MB
  __bf16* ybuf = vtb + (size_t)BATCH * HEADS * HS * SEQ;  // 16 MB

  dim3 blk(256);

  f32_to_bf16<<<(M * EMBED) / (256 * 8), blk, 0, stream>>>(x, xb, M * EMBED);
  f32_to_bf16<<<(C3 * EMBED) / (256 * 8), blk, 0, stream>>>(w_attn, wab,
                                                            C3 * EMBED);
  f32_to_bf16<<<(EMBED * EMBED) / (256 * 8), blk, 0, stream>>>(w_proj, wpb,
                                                               EMBED * EMBED);

  gemm_bias_wmma<__bf16><<<dim3(C3 / 256, M / 64), blk, 0, stream>>>(
      xb, wab, b_attn, qkvb, C3, EMBED);

  transpose_v<<<(BATCH * HEADS * HS * SEQ) / 256, blk, 0, stream>>>(qkvb, vtb);

  attn_wmma<<<dim3(SEQ / 128, HEADS, BATCH), blk, 0, stream>>>(qkvb, vtb, ybuf);

  gemm_bias_wmma<float><<<dim3(EMBED / 256, M / 64), blk, 0, stream>>>(
      ybuf, wpb, b_proj, out, EMBED, EMBED);
}
